// CustomMultiheadAttention_40441412059242
// MI455X (gfx1250) — compile-verified, matrix operand order checked by reading, not run on hardware
//
#include <hip/hip_runtime.h>

// Problem constants (match reference)
#define Bn 4
#define Tn 1024
#define Sn 1024
#define En 1024
#define Hn 16
#define Dn 64

typedef __attribute__((ext_vector_type(16))) _Float16 v16h;
typedef __attribute__((ext_vector_type(8)))  _Float16 h8;
typedef __attribute__((ext_vector_type(8)))  float    v8f;
typedef __attribute__((ext_vector_type(4)))  int      v4i;

#if __has_builtin(__builtin_amdgcn_global_load_async_to_lds_b128)
#define USE_ASYNC_LDS 1
#else
#define USE_ASYNC_LDS 0
#endif

#if USE_ASYNC_LDS
typedef __attribute__((address_space(1))) v4i as1_v4i;  // global
typedef __attribute__((address_space(3))) v4i as3_v4i;  // LDS
// Copy one 16-byte chunk global -> LDS via the async path (ASYNCcnt).
__device__ __forceinline__ void async_cp16(const _Float16* g, _Float16* l) {
  __builtin_amdgcn_global_load_async_to_lds_b128((as1_v4i*)g, (as3_v4i*)l, 0,
                                                 0);
}
__device__ __forceinline__ void wait_async0() {
#if __has_builtin(__builtin_amdgcn_s_wait_asynccnt)
  __builtin_amdgcn_s_wait_asynccnt(0);
#else
  asm volatile("s_wait_asynccnt 0" ::: "memory");
#endif
}
#endif

__device__ __forceinline__ v8f wmma16(v16h a, v16h b, v8f c) {
  // D = A(16x32 f16) * B(32x16 f16) + C(16x16 f32)
  return __builtin_amdgcn_wmma_f32_16x16x32_f16(false, a, false, b, (short)0, c,
                                                false, false);
}

// Load one 16-half WMMA operand fragment from an LDS tile.
// Per ISA layout, each lane needs K = g*16 + hi*8 + j (j=0..7, g=0..1), i.e.
// two contiguous 16-byte chunks at (off) and (off+16).
__device__ __forceinline__ v16h ld_frag(const _Float16* base, int off) {
  union { v16h v; h8 h[2]; } u;
  u.h[0] = *(const h8*)(base + off);
  u.h[1] = *(const h8*)(base + off + 16);
  return u.v;
}

// ---------------------------------------------------------------------------
// Wout f32 -> f16 (layout kept [n][k] row-major == Wout itself, since the
// GEMM needs B[k][n] = Wout[n][k])
// ---------------------------------------------------------------------------
__global__ void cvt_kernel(const float* __restrict__ src,
                           _Float16* __restrict__ dst, int nelem) {
  int i = blockIdx.x * blockDim.x + threadIdx.x;
  if (i < nelem) dst[i] = (_Float16)src[i];
}

// ---------------------------------------------------------------------------
// Q/K/V projection: X[N,64] @ W^T[64,64] + b -> f16 with head-major row remap.
// mode 0: rows are (b,t,h) -> out row (b*H+h)*T + t   (query)
// mode 1: rows are (s,h)   -> out row h*S + s         (key/value)
// 128 threads (4 waves). Each wave: 16 rows x 64 cols = 4 col-tiles, K=64.
// ---------------------------------------------------------------------------
__global__ __launch_bounds__(128) void proj_kernel(
    const float* __restrict__ X, const float* __restrict__ W,
    const float* __restrict__ bias, _Float16* __restrict__ out, int mode) {
  __shared__ __align__(16) _Float16 w_lds[64 * 64];      // [n][k]
  __shared__ __align__(16) _Float16 x_lds[4][16 * 64];   // per wave [m][k]

  const int tid  = threadIdx.x;
  const int wave = tid >> 5, lane = tid & 31;
  const int n = lane & 15, hi = lane >> 4;
  const int rbase = blockIdx.x * 64 + wave * 16;

  for (int j = 0; j < 32; ++j) {
    int idx = tid + j * 128;
    w_lds[idx] = (_Float16)W[idx];   // B[k][n] = W[n][k] -> store W row-major
  }
  _Float16* xw = x_lds[wave];
  for (int j = 0; j < 32; ++j) {
    int idx = lane + j * 32;
    int m = idx >> 6, c = idx & 63;
    xw[m * 64 + c] = (_Float16)X[(size_t)(rbase + m) * 64 + c];
  }
  __syncthreads();

  v16h a0 = ld_frag(xw, n * 64 + 0  + hi * 8);
  v16h a1 = ld_frag(xw, n * 64 + 32 + hi * 8);

  v8f acc[4];
  for (int ct = 0; ct < 4; ++ct) {
    v8f c = {};
    v16h b0 = ld_frag(w_lds, (ct * 16 + n) * 64 + 0  + hi * 8);
    v16h b1 = ld_frag(w_lds, (ct * 16 + n) * 64 + 32 + hi * 8);
    c = wmma16(a0, b0, c);
    c = wmma16(a1, b1, c);
    acc[ct] = c;
  }

  for (int ct = 0; ct < 4; ++ct) {
    float bv = bias[ct * 16 + n];
    for (int r = 0; r < 8; ++r) {
      int grow = rbase + hi * 8 + r;
      int orow;
      if (mode == 0) {
        int bb  = grow >> 14;        // / (T*H) = 16384
        int rem = grow & 16383;
        int t = rem >> 4, h = rem & 15;
        orow = (bb * Hn + h) * Tn + t;
      } else {
        int s = grow >> 4, h = grow & 15;
        orow = h * Sn + s;
      }
      out[(size_t)orow * 64 + ct * 16 + n] = (_Float16)(acc[ct][r] + bv);
    }
  }
}

// ---------------------------------------------------------------------------
// Flash attention: one workgroup = 64 query rows of one (b,h); 4 waves,
// each wave owns 16 rows. Online softmax over S in chunks of 32.
// Per chunk per wave: 4 WMMAs for scores, 4 WMMAs for P@V.
// ---------------------------------------------------------------------------
__global__ __launch_bounds__(128) void attn_kernel(
    const _Float16* __restrict__ Qp, const _Float16* __restrict__ Kp,
    const _Float16* __restrict__ Vp, _Float16* __restrict__ Ctx) {
  __shared__ __align__(16) _Float16 q_lds[4][16 * 64];  // per wave [m][d]
  __shared__ __align__(16) _Float16 k_lds[32 * 64];     // [s][d] (=B [n][k])
  __shared__ __align__(16) _Float16 v_lds[64 * 32];     // [d][s] (=B [n][k])
  __shared__ __align__(16) _Float16 p_lds[4][16 * 32];  // per wave [m][s]

  const int tid  = threadIdx.x;
  const int wave = tid >> 5, lane = tid & 31;
  const int n = lane & 15, hi = lane >> 4;
  const int tt = blockIdx.x & 15;
  const int bh = blockIdx.x >> 4;
  const int b = bh >> 4, h = bh & 15;
  const int t0 = tt * 64 + wave * 16;

  // stage this wave's 16x64 Q tile (2 KB = 128 x 16B chunks, 4 per lane)
  _Float16* qw = q_lds[wave];
  {
    const _Float16* qs = Qp + ((size_t)(b * Hn + h) * Tn + t0) * 64;
#if USE_ASYNC_LDS
    for (int j = 0; j < 4; ++j) {
      int c = lane + j * 32;
      async_cp16(qs + c * 8, qw + c * 8);
    }
    wait_async0();
#else
    const unsigned int* qu = (const unsigned int*)qs;
    unsigned int* qd = (unsigned int*)qw;
    for (int j = 0; j < 16; ++j) qd[lane + j * 32] = qu[lane + j * 32];
#endif
  }
  __syncthreads();

  v16h aq0 = ld_frag(qw, n * 64 + 0  + hi * 8);
  v16h aq1 = ld_frag(qw, n * 64 + 32 + hi * 8);

  v8f o0 = {}, o1 = {}, o2 = {}, o3 = {};
  float mrow[8], lrow[8];
  for (int r = 0; r < 8; ++r) { mrow[r] = -3.0e38f; lrow[r] = 0.0f; }
  const float sc = 0.03125f;  // 1/sqrt(E) = 1/32 (reference scales by sqrt(E))

  const size_t kvbase = (size_t)h * Sn * 64;
  for (int s0 = 0; s0 < Sn; s0 += 32) {
    __syncthreads();  // prior chunk's K/V reads complete
    {   // stage K chunk [32 s][64 d]: 4 KB = 256 x 16B chunks, 2 per thread
      const _Float16* ks = Kp + kvbase + (size_t)s0 * 64;
#if USE_ASYNC_LDS
      for (int j = 0; j < 2; ++j) {
        int c = tid + j * 128;
        async_cp16(ks + c * 8, k_lds + c * 8);
      }
#else
      unsigned int* kd = (unsigned int*)k_lds;
      const unsigned int* ku = (const unsigned int*)ks;
      for (int j = 0; j < 8; ++j) kd[tid + j * 128] = ku[tid + j * 128];
#endif
    }
    {   // stage V chunk transposed -> [64 d][32 s] (scatter: VALU path)
      const _Float16* vs = Vp + kvbase + (size_t)s0 * 64;
      for (int j = 0; j < 16; ++j) {
        int idx = tid + j * 128;
        int d = idx >> 5, s = idx & 31;
        v_lds[d * 32 + s] = vs[s * 64 + d];
      }
    }
#if USE_ASYNC_LDS
    wait_async0();
#endif
    __syncthreads();

    // scores: two 16x16 col-tiles, K=64 over D in two 32-steps
    v8f sf0 = {}, sf1 = {};
    {
      v16h bk;
      bk = ld_frag(k_lds, (0 * 16 + n) * 64 + 0  + hi * 8); sf0 = wmma16(aq0, bk, sf0);
      bk = ld_frag(k_lds, (0 * 16 + n) * 64 + 32 + hi * 8); sf0 = wmma16(aq1, bk, sf0);
      bk = ld_frag(k_lds, (1 * 16 + n) * 64 + 0  + hi * 8); sf1 = wmma16(aq0, bk, sf1);
      bk = ld_frag(k_lds, (1 * 16 + n) * 64 + 32 + hi * 8); sf1 = wmma16(aq1, bk, sf1);
    }

    // online softmax (row r lives in the lane's 16-lane half at reg index r)
    float p0[8], p1[8], mnew[8], alpha[8], rsum[8];
    for (int r = 0; r < 8; ++r) {
      p0[r] = sf0[r] * sc;
      p1[r] = sf1[r] * sc;
      mnew[r] = fmaxf(p0[r], p1[r]);
    }
    for (int off = 1; off <= 8; off <<= 1)
      for (int r = 0; r < 8; ++r)
        mnew[r] = fmaxf(mnew[r], __shfl_xor(mnew[r], off, 32));
    for (int r = 0; r < 8; ++r) {
      mnew[r]  = fmaxf(mnew[r], mrow[r]);
      alpha[r] = __expf(mrow[r] - mnew[r]);
      mrow[r]  = mnew[r];
      p0[r] = __expf(p0[r] - mnew[r]);
      p1[r] = __expf(p1[r] - mnew[r]);
      rsum[r] = p0[r] + p1[r];
    }
    for (int off = 1; off <= 8; off <<= 1)
      for (int r = 0; r < 8; ++r)
        rsum[r] += __shfl_xor(rsum[r], off, 32);
    for (int r = 0; r < 8; ++r) {
      lrow[r] = lrow[r] * alpha[r] + rsum[r];
      o0[r] *= alpha[r]; o1[r] *= alpha[r];
      o2[r] *= alpha[r]; o3[r] *= alpha[r];
    }

    // re-layout P: C/D accumulator layout -> A fragment layout via LDS
    _Float16* pw = p_lds[wave];
    for (int r = 0; r < 8; ++r) {
      pw[(hi * 8 + r) * 32 + n]      = (_Float16)p0[r];
      pw[(hi * 8 + r) * 32 + 16 + n] = (_Float16)p1[r];
    }
    __syncthreads();

    v16h pa = ld_frag(pw, n * 32 + hi * 8);
    v16h bv;
    bv = ld_frag(v_lds, (0 * 16 + n) * 32 + hi * 8); o0 = wmma16(pa, bv, o0);
    bv = ld_frag(v_lds, (1 * 16 + n) * 32 + hi * 8); o1 = wmma16(pa, bv, o1);
    bv = ld_frag(v_lds, (2 * 16 + n) * 32 + hi * 8); o2 = wmma16(pa, bv, o2);
    bv = ld_frag(v_lds, (3 * 16 + n) * 32 + hi * 8); o3 = wmma16(pa, bv, o3);
  }

  // normalize and write context in [B,T,E] (f16) for the output GEMM
  for (int r = 0; r < 8; ++r) {
    float inv = 1.0f / lrow[r];
    int t = t0 + hi * 8 + r;
    size_t rowoff = ((size_t)b * Tn + t) * En + h * 64;
    Ctx[rowoff + 0 * 16 + n] = (_Float16)(o0[r] * inv);
    Ctx[rowoff + 1 * 16 + n] = (_Float16)(o1[r] * inv);
    Ctx[rowoff + 2 * 16 + n] = (_Float16)(o2[r] * inv);
    Ctx[rowoff + 3 * 16 + n] = (_Float16)(o3[r] * inv);
  }
}

// ---------------------------------------------------------------------------
// Output projection: Ctx[4096,1024] @ Wout^T + bout -> f32 out.
// 256 threads (8 waves) per WG compute a 128x64 tile; K loop in steps of 32.
// Each wave = one 16-row stripe x 4 col sub-tiles: 1 A-frag reused over 4
// B-frags -> 4 WMMAs per wave per k-step (32 per WG per 12KB staged).
// ---------------------------------------------------------------------------
__global__ __launch_bounds__(256) void outproj_kernel(
    const _Float16* __restrict__ Ctx, const _Float16* __restrict__ Wf,
    const float* __restrict__ bout, float* __restrict__ Out) {
  __shared__ __align__(16) _Float16 a_lds[128 * 32];  // [m][k]
  __shared__ __align__(16) _Float16 b_lds[64 * 32];   // [n][k]

  const int tid  = threadIdx.x;
  const int wave = tid >> 5, lane = tid & 31;
  const int n = lane & 15, hi = lane >> 4;
  const int rb = blockIdx.x >> 4, cb = blockIdx.x & 15;
  const int r0 = rb * 128, n0 = cb * 64;

  v8f acc[4] = {{}, {}, {}, {}};

  for (int k0 = 0; k0 < En; k0 += 32) {
    __syncthreads();
#if USE_ASYNC_LDS
    // A: 8 KB = 512 x 16B chunks (2/thread); B: 4 KB = 256 x 16B (1/thread)
    for (int j = 0; j < 2; ++j) {
      int ca = tid + j * 256;
      int row = ca >> 2, sub = ca & 3;
      async_cp16(Ctx + (size_t)(r0 + row) * En + k0 + sub * 8, a_lds + ca * 8);
    }
    {
      int cbk = tid;
      int row = cbk >> 2, sub = cbk & 3;
      async_cp16(Wf + (size_t)(n0 + row) * En + k0 + sub * 8, b_lds + cbk * 8);
    }
    wait_async0();
#else
    {
      unsigned int* ad = (unsigned int*)a_lds;
      unsigned int* bd = (unsigned int*)b_lds;
      const unsigned int* cu = (const unsigned int*)Ctx;
      const unsigned int* wu = (const unsigned int*)Wf;
      for (int j = 0; j < 8; ++j) {
        int u = tid + j * 256;
        int row = u >> 4, c = u & 15;
        ad[row * 16 + c] = cu[(size_t)(r0 + row) * 512 + (k0 >> 1) + c];
      }
      for (int j = 0; j < 4; ++j) {
        int u = tid + j * 256;
        int row = u >> 4, c = u & 15;
        bd[row * 16 + c] = wu[(size_t)(n0 + row) * 512 + (k0 >> 1) + c];
      }
    }
#endif
    if (k0 + 32 < En) {  // prefetch next k-tile into L2/L0
      int row = tid >> 1, sub = tid & 1;
      __builtin_prefetch(Ctx + (size_t)(r0 + row) * En + k0 + 32 + sub * 16, 0, 0);
      if (tid < 128)
        __builtin_prefetch(Wf + (size_t)(n0 + (tid >> 1)) * En + k0 + 32 +
                               (tid & 1) * 16, 0, 0);
    }
    __syncthreads();

    v16h a = ld_frag(a_lds, (wave * 16 + n) * 32 + hi * 8);
    for (int ct = 0; ct < 4; ++ct) {
      v16h b = ld_frag(b_lds, (ct * 16 + n) * 32 + hi * 8);
      acc[ct] = wmma16(a, b, acc[ct]);
    }
  }

  for (int ct = 0; ct < 4; ++ct) {
    float bias = bout[n0 + ct * 16 + n];
    for (int r = 0; r < 8; ++r) {
      int row = r0 + wave * 16 + hi * 8 + r;
      Out[(size_t)row * En + n0 + ct * 16 + n] = acc[ct][r] + bias;
    }
  }
}

// ---------------------------------------------------------------------------
extern "C" void kernel_launch(void* const* d_in, const int* in_sizes, int n_in,
                              void* d_out, int out_size, void* d_ws,
                              size_t ws_size, hipStream_t stream) {
  (void)in_sizes; (void)n_in; (void)out_size; (void)ws_size;
  const float* query = (const float*)d_in[0];
  const float* para  = (const float*)d_in[1];
  const float* Wq = (const float*)d_in[2];
  const float* bq = (const float*)d_in[3];
  const float* Wk = (const float*)d_in[4];
  const float* bk = (const float*)d_in[5];
  const float* Wv = (const float*)d_in[6];
  const float* bv = (const float*)d_in[7];
  const float* Wout = (const float*)d_in[8];
  const float* bout = (const float*)d_in[9];
  float* out = (float*)d_out;

  // workspace layout (all regions fully overwritten before being read)
  char* ws = (char*)d_ws;
  _Float16* qp  = (_Float16*)(ws);                       // 8 MB: [B*H, T, 64]
  _Float16* kp  = (_Float16*)(ws + (size_t)(8u  << 20)); // 2 MB: [H, S, 64]
  _Float16* vp  = (_Float16*)(ws + (size_t)(10u << 20)); // 2 MB: [H, S, 64]
  _Float16* ctx = (_Float16*)(ws + (size_t)(12u << 20)); // 8 MB: [B*T, E]
  _Float16* wf  = (_Float16*)(ws + (size_t)(20u << 20)); // 2 MB: Wout f16

  hipLaunchKernelGGL(cvt_kernel, dim3(4096), dim3(256), 0, stream,
                     Wout, wf, En * En);
  hipLaunchKernelGGL(proj_kernel, dim3(1024), dim3(128), 0, stream,
                     query, Wq, bq, qp, 0);
  hipLaunchKernelGGL(proj_kernel, dim3(256), dim3(128), 0, stream,
                     para, Wk, bk, kp, 1);
  hipLaunchKernelGGL(proj_kernel, dim3(256), dim3(128), 0, stream,
                     para, Wv, bv, vp, 1);
  hipLaunchKernelGGL(attn_kernel, dim3(1024), dim3(128), 0, stream,
                     qp, kp, vp, ctx);
  hipLaunchKernelGGL(outproj_kernel, dim3(512), dim3(256), 0, stream,
                     ctx, wf, bout, out);
}
